// Model_8521215115887
// MI455X (gfx1250) — matrix operand order repaired and checked by reading, not compiled, not run on hardware
//
#include <hip/hip_runtime.h>
#include <hip/hip_bf16.h>

typedef __attribute__((ext_vector_type(16))) _Float16 v16h;
typedef __attribute__((ext_vector_type(8)))  float    v8f;

union AF { v16h h; unsigned u[8]; };

#define FLAG_ACC   1
#define FLAG_RELU  2
#define FLAG_LRELU 4

constexpr int ND = 50000, NP = 20000, EE = 1000000, EL = 250000;
constexpr float BN_INV = 0.9999950000374997f;

__device__ inline unsigned pk2(float lo, float hi) {
  union { _Float16 h[2]; unsigned u; } q;
  q.h[0] = (_Float16)lo; q.h[1] = (_Float16)hi;
  return q.u;
}

// ---------------------------------------------------------------------------
// Generic WMMA GEMM: C[M,N] = act(A[M,K] @ W[K,N] + bias (+gather) (+C))
// f32 global --(float4 loads, f16-pair packed)--> LDS --> v_wmma_f32_16x16x32_f16
// Block = 256 thr = 8 waves; tile 32 rows x 64 cols; software-pipelined tiles.
// Requires: K % 32 == 0, N % 64 == 0 (true for every call site).
// ---------------------------------------------------------------------------
__global__ __launch_bounds__(256) void gemm_wmma_f16(
    const float* __restrict__ A, const float* __restrict__ W,
    const float* __restrict__ bias, float* __restrict__ C,
    const int* __restrict__ gidx, const float* __restrict__ gtab,
    int M, int N, int K, int flags)
{
  // K-pair-packed dword tiles: As32[row][kpair], Bs32[kpair][col]
  __shared__ unsigned As32[32][18];   // 32 rows x 16 kpairs (+2 pad)
  __shared__ unsigned Bs32[16][68];   // 16 kpairs x 64 cols (+4 pad)
  const int t = threadIdx.x;
  const int lane = t & 31, wave = t >> 5;
  const int rowBase = blockIdx.x * 32;
  const int colBase = blockIdx.y * 64;
  const int wr = (wave >> 2) * 16;
  const int wc = (wave & 3) * 16;
  const int r  = wr + (lane & 15);
  const int nn = wc + (lane & 15);
  const int a4 = (lane >= 16) ? 4 : 0;
  const int b8 = (lane >= 16) ? 8 : 0;

  // staging coordinates (guard-free: clamp rows, N always multiple of 64)
  const int sr  = t >> 3;          // A row 0..31
  const int sk4 = (t & 7) * 4;     // A col chunk
  int gr = rowBase + sr; if (gr >= M) gr = M - 1;
  const float* Aptr = A + (long)gr * K + sk4;
  const int wkp = t >> 4;          // W kpair 0..15
  const int wn4 = (t & 15) * 4;    // W col chunk
  const float* Wptr = W + (long)(2 * wkp) * N + colBase + wn4;

  float4 aR, w0R, w1R;
  auto ldg = [&](int kb) {
    aR  = *reinterpret_cast<const float4*>(Aptr + kb);
    w0R = *reinterpret_cast<const float4*>(Wptr + (long)kb * N);
    w1R = *reinterpret_cast<const float4*>(Wptr + (long)kb * N + N);
  };
  auto sts = [&]() {
    *reinterpret_cast<uint2*>(&As32[sr][sk4 >> 1]) =
        make_uint2(pk2(aR.x, aR.y), pk2(aR.z, aR.w));
    *reinterpret_cast<uint4*>(&Bs32[wkp][wn4]) =
        make_uint4(pk2(w0R.x, w1R.x), pk2(w0R.y, w1R.y),
                   pk2(w0R.z, w1R.z), pk2(w0R.w, w1R.w));
  };

  v8f acc = {};
  ldg(0); sts();
  for (int kb = 0; kb < K; kb += 32) {
    __syncthreads();
    const bool more = (kb + 32) < K;
    if (more) ldg(kb + 32);                       // overlap with MMA below
    if (kb + 64 < K) __builtin_prefetch(Aptr + kb + 64, 0, 1);
    AF a, b;
    #pragma unroll
    for (int vv = 0; vv < 8; ++vv) {
      a.u[vv] = As32[r][(vv >> 2) * 8 + a4 + (vv & 3)];  // ISA A-frag layout
      b.u[vv] = Bs32[(vv >> 2) * 8 + a4 + (vv & 3)][nn]; // ISA B-frag layout
    }
    acc = __builtin_amdgcn_wmma_f32_16x16x32_f16(false, a.h, false, b.h,
                                                 (short)0, acc, false, false);
    __syncthreads();
    if (more) sts();
  }

  const int col  = colBase + wc + (lane & 15);
  const int rofs = rowBase + wr + ((lane >= 16) ? 8 : 0);
  #pragma unroll
  for (int i = 0; i < 8; ++i) {
    int row = rofs + i;
    if (row < M && col < N) {
      float v = acc[i];
      if (bias) v += bias[col];
      if (gtab) v += gtab[(long)gidx[row] * N + col];
      long o = (long)row * N + col;
      if (flags & FLAG_ACC)   v += C[o];
      if (flags & FLAG_RELU)  v = fmaxf(v, 0.f);
      if (flags & FLAG_LRELU) v = (v > 0.f) ? v : 0.01f * v;
      C[o] = v;
    }
  }
}

// ---------------------------------------------------------------------------
// Graph scatter kernels (wave32 per edge)
// ---------------------------------------------------------------------------
__global__ __launch_bounds__(256) void sage_agg_k(
    const float* __restrict__ x, const int* __restrict__ src,
    const int* __restrict__ dst, int E_, float* __restrict__ agg,
    float* __restrict__ deg)
{
  int wave = threadIdx.x >> 5, lane = threadIdx.x & 31;
  int e = blockIdx.x * 8 + wave;
  if (e >= E_) return;
  int s = src[e], d = dst[e];
  atomicAdd(&agg[(long)d * 64 + lane],      x[(long)s * 64 + lane]);
  atomicAdd(&agg[(long)d * 64 + lane + 32], x[(long)s * 64 + lane + 32]);
  if (lane == 0) atomicAdd(&deg[d], 1.0f);
}

__global__ void sage_mean_k(float* agg, const float* deg, int total)
{
  int i = blockIdx.x * 256 + threadIdx.x;
  if (i >= total) return;
  agg[i] = agg[i] / fmaxf(deg[i >> 6], 1.0f);
}

__device__ inline unsigned fmap(float f) {
  unsigned u = __float_as_uint(f);
  return (u & 0x80000000u) ? ~u : (u | 0x80000000u);
}
__device__ inline float funmap(unsigned u) {
  return (u & 0x80000000u) ? __uint_as_float(u ^ 0x80000000u)
                           : __uint_as_float(~u);
}

__global__ void att_reduce_k(const float* __restrict__ h,
                             const float* __restrict__ a,
                             float* __restrict__ al, int total, int hshift, int ch)
{
  int i = blockIdx.x * 256 + threadIdx.x;
  if (i >= total) return;
  int hh = i & ((1 << hshift) - 1), n = i >> hshift;
  float s = 0.f;
  for (int c = 0; c < ch; ++c) s += h[(long)n * 64 + hh * ch + c] * a[hh * ch + c];
  al[i] = s;
}

__global__ void edge_max_k(const float* __restrict__ als,
                           const float* __restrict__ ald,
                           const int* __restrict__ src, const int* __restrict__ dst,
                           int E_, int hshift, float* __restrict__ ebuf,
                           unsigned* __restrict__ m)
{
  int i = blockIdx.x * 256 + threadIdx.x;
  if (i >= (E_ << hshift)) return;
  int hh = i & ((1 << hshift) - 1), e = i >> hshift;
  int s = src[e], d = dst[e];
  float v = als[(s << hshift) + hh] + ald[(d << hshift) + hh];
  v = (v > 0.f) ? v : 0.2f * v;                 // leaky_relu(0.2)
  ebuf[i] = v;
  atomicMax(&m[(d << hshift) + hh], fmap(v));
}

__global__ void edge_expsum_k(float* __restrict__ ebuf,
                              const unsigned* __restrict__ m,
                              float* __restrict__ den,
                              const int* __restrict__ dst, int E_, int hshift)
{
  int i = blockIdx.x * 256 + threadIdx.x;
  if (i >= (E_ << hshift)) return;
  int hh = i & ((1 << hshift) - 1), e = i >> hshift;
  int d = dst[e];
  float mv = funmap(m[(d << hshift) + hh]);
  if (!(fabsf(mv) <= 1e38f)) mv = 0.f;
  float ex = expf(ebuf[i] - mv);
  ebuf[i] = ex;
  atomicAdd(&den[(d << hshift) + hh], ex);
}

__global__ __launch_bounds__(256) void edge_agg_k(
    const float* __restrict__ hs, const float* __restrict__ ebuf,
    const float* __restrict__ den, const int* __restrict__ src,
    const int* __restrict__ dst, int E_, int hshift, int cshift,
    float* __restrict__ gout)
{
  int wave = threadIdx.x >> 5, lane = threadIdx.x & 31;
  int e = blockIdx.x * 8 + wave;
  if (e >= E_) return;
  int s = src[e], d = dst[e];
  #pragma unroll
  for (int rep = 0; rep < 2; ++rep) {
    int c = lane + rep * 32;
    int hh = c >> cshift;
    float alpha = ebuf[((long)e << hshift) + hh] /
                  fmaxf(den[(d << hshift) + hh], 1e-16f);
    atomicAdd(&gout[(long)d * 64 + c], alpha * hs[(long)s * 64 + c]);
  }
}

// ---------------------------------------------------------------------------
// Elementwise kernels
// ---------------------------------------------------------------------------
__global__ void combine_k(float* __restrict__ x, const float* __restrict__ sraw,
                          const float* __restrict__ gsum,
                          const float* __restrict__ gb, int total)
{
  int i = blockIdx.x * 256 + threadIdx.x;
  if (i >= total) return;
  float s = fmaxf(sraw[i], 0.f);
  float g = fmaxf(gsum[i] + gb[i & 63], 0.f);
  x[i] += s * (1.f / (1.f + expf(-s))) + g * (1.f / (1.f + expf(-g)));
}

__global__ void mix_k(float* a, const float* b, int total)
{
  int i = blockIdx.x * 256 + threadIdx.x;
  if (i >= total) return;
  a[i] = 0.5f * (a[i] + b[i]);
}

__global__ void maxpool_k(const float* in, float* out, int total)
{
  int i = blockIdx.x * 256 + threadIdx.x;   // total = NP*32
  if (i >= total) return;
  int n = i >> 5, j = i & 31;
  out[i] = fmaxf(in[(long)n * 64 + 2 * j], in[(long)n * 64 + 2 * j + 1]);
}

__global__ void bn_lrelu_k(float* buf, const float* g, const float* b,
                           int total, int C)
{
  int i = blockIdx.x * 256 + threadIdx.x;
  if (i >= total) return;
  int c = i % C;     // C is 64/128/256 -> strength-reduced to mask by compiler
  float v = buf[i];
  v = (v > 0.f) ? v : 0.01f * v;
  buf[i] = v * (g[c] * BN_INV) + b[c];
}

__global__ void sigmoid_k(float* buf, int total)
{
  int i = blockIdx.x * 256 + threadIdx.x;
  if (i >= total) return;
  buf[i] = 1.f / (1.f + expf(-buf[i]));
}

// ---------------------------------------------------------------------------
// Fused classifier: gather-concat -> WMMA GEMM(128->64) -> ReLU -> dot(64->1)
// ---------------------------------------------------------------------------
__global__ __launch_bounds__(256) void classify_k(
    const float* __restrict__ xd, const float* __restrict__ p,
    const int* __restrict__ lsrc, const int* __restrict__ ldst,
    const float* __restrict__ W1, const float* __restrict__ b1,
    const float* __restrict__ w2, const float* __restrict__ b2,
    float* __restrict__ out, int Elbl)
{
  __shared__ unsigned EF32[32][66];   // 32 rows x 64 kpairs (+2 pad)
  __shared__ unsigned WT32[64][68];   // 64 kpairs x 64 cols (+4 pad)
  __shared__ float HS[32][64];
  __shared__ float PART[32][8];
  __shared__ int SI[32], DI[32];
  const int t = threadIdx.x;
  const int lane = t & 31, wave = t >> 5;
  const int rowBase = blockIdx.x * 32;

  if (t < 32) {
    int gr = rowBase + t;
    SI[t] = (gr < Elbl) ? lsrc[gr] : 0;
    DI[t] = (gr < Elbl) ? ldst[gr] : 0;
  }
  #pragma unroll
  for (int i = 0; i < 4; ++i) {            // stage W1 as kpair-packed dwords
    int id = t + 256 * i;
    int kp = id >> 4, n4 = (id & 15) * 4;
    float4 a0 = *reinterpret_cast<const float4*>(W1 + (2 * kp) * 64 + n4);
    float4 a1 = *reinterpret_cast<const float4*>(W1 + (2 * kp + 1) * 64 + n4);
    *reinterpret_cast<uint4*>(&WT32[kp][n4]) =
        make_uint4(pk2(a0.x, a1.x), pk2(a0.y, a1.y),
                   pk2(a0.z, a1.z), pk2(a0.w, a1.w));
  }
  __syncthreads();
  #pragma unroll
  for (int i = 0; i < 4; ++i) {            // gather-concat ef rows (float4)
    int id = t + 256 * i;
    int r = id >> 5, c4 = (id & 31) * 4;
    float4 v = (c4 < 64)
        ? *reinterpret_cast<const float4*>(xd + (long)SI[r] * 64 + c4)
        : *reinterpret_cast<const float4*>(p + (long)DI[r] * 64 + (c4 - 64));
    *reinterpret_cast<uint2*>(&EF32[r][c4 >> 1]) =
        make_uint2(pk2(v.x, v.y), pk2(v.z, v.w));
  }
  __syncthreads();

  const int wr = (wave >> 2) * 16, wc = (wave & 3) * 16;
  const int r  = wr + (lane & 15);
  const int nn = wc + (lane & 15);
  const int a4 = (lane >= 16) ? 4 : 0;
  const int b8 = (lane >= 16) ? 8 : 0;
  v8f acc = {};
  #pragma unroll
  for (int kb = 0; kb < 128; kb += 32) {
    const int kd0 = kb >> 1;
    AF a, b;
    #pragma unroll
    for (int vv = 0; vv < 8; ++vv) {
      a.u[vv] = EF32[r][kd0 + (vv >> 2) * 8 + a4 + (vv & 3)];
      b.u[vv] = WT32[kd0 + (vv >> 2) * 8 + a4 + (vv & 3)][nn];
    }
    acc = __builtin_amdgcn_wmma_f32_16x16x32_f16(false, a.h, false, b.h,
                                                 (short)0, acc, false, false);
  }
  const int colL = wc + (lane & 15);
  const int rowL = wr + ((lane >= 16) ? 8 : 0);
  #pragma unroll
  for (int i = 0; i < 8; ++i)
    HS[rowL + i][colL] = fmaxf(acc[i] + b1[colL], 0.f);
  __syncthreads();

  int rr = t >> 3, pp = t & 7;
  float s = 0.f;
  #pragma unroll
  for (int j = 0; j < 8; ++j) s += HS[rr][pp * 8 + j] * w2[pp * 8 + j];
  PART[rr][pp] = s;
  __syncthreads();
  if (t < 32) {
    float a2 = b2[0];
    #pragma unroll
    for (int j = 0; j < 8; ++j) a2 += PART[t][j];
    int gr = rowBase + t;
    if (gr < Elbl) out[gr] = a2;
  }
}

// ---------------------------------------------------------------------------
// Host-side orchestration
// ---------------------------------------------------------------------------
#define GEMM(A, W, B, C, GI, GT, M, N, K, F)                                   \
  gemm_wmma_f16<<<dim3(((M) + 31) / 32, ((N) + 63) / 64), 256, 0, stream>>>(   \
      (A), (W), (B), (C), (GI), (GT), (M), (N), (K), (F))
#define EWG(n) <<<(((n) + 255) / 256), 256, 0, stream>>>

extern "C" void kernel_launch(void* const* d_in, const int* in_sizes, int n_in,
                              void* d_out, int out_size, void* d_ws, size_t ws_size,
                              hipStream_t stream)
{
  const float *x_drug = (const float*)d_in[0],  *x_prot = (const float*)d_in[1];
  const float *W_dl = (const float*)d_in[2],    *b_dl = (const float*)d_in[3];
  const float *W_pl = (const float*)d_in[4],    *b_pl = (const float*)d_in[5];
  const float *emb_drug = (const float*)d_in[6],*emb_prot = (const float*)d_in[7];
  const float *s1_Wl = (const float*)d_in[8], *s1_bl = (const float*)d_in[9], *s1_Wr = (const float*)d_in[10];
  const float *s2_Wl = (const float*)d_in[11],*s2_bl = (const float*)d_in[12],*s2_Wr = (const float*)d_in[13];
  const float *g1_W = (const float*)d_in[14], *g1_as = (const float*)d_in[15], *g1_ad = (const float*)d_in[16], *g1_b = (const float*)d_in[17];
  const float *g2_W = (const float*)d_in[18], *g2_as = (const float*)d_in[19], *g2_ad = (const float*)d_in[20], *g2_b = (const float*)d_in[21];
  const float *enc_W = (const float*)d_in[22], *enc_b = (const float*)d_in[23];
  const float *mha_Wv = (const float*)d_in[24], *mha_bv = (const float*)d_in[25];
  const float *mha_Wo = (const float*)d_in[26], *mha_bo = (const float*)d_in[27];
  const float *h_W1 = (const float*)d_in[28], *h_b1 = (const float*)d_in[29];
  const float *h_W2 = (const float*)d_in[30], *h_b2 = (const float*)d_in[31];
  const float *h_W3 = (const float*)d_in[32], *h_b3 = (const float*)d_in[33];
  const float *o1_W = (const float*)d_in[34], *o1_b = (const float*)d_in[35];
  const float *bn1_g = (const float*)d_in[36], *bn1_b = (const float*)d_in[37];
  const float *o2_W = (const float*)d_in[38], *o2_b = (const float*)d_in[39];
  const float *bn2_g = (const float*)d_in[40], *bn2_b = (const float*)d_in[41];
  const float *o3_W = (const float*)d_in[42], *o3_b = (const float*)d_in[43];
  const float *bn3_g = (const float*)d_in[44], *bn3_b = (const float*)d_in[45];
  const float *o4_W = (const float*)d_in[46], *o4_b = (const float*)d_in[47];
  const float *c_W1 = (const float*)d_in[48], *c_b1 = (const float*)d_in[49];
  const float *c_W2 = (const float*)d_in[50], *c_b2 = (const float*)d_in[51];
  const int *drug_id = (const int*)d_in[52], *prot_id = (const int*)d_in[53];
  const int *e_src = (const int*)d_in[54], *e_dst = (const int*)d_in[55];
  const int *l_src = (const int*)d_in[56], *l_dst = (const int*)d_in[57];
  float* out = (float*)d_out;
  (void)in_sizes; (void)n_in; (void)out_size; (void)ws_size;

  char* wp = (char*)d_ws;
  auto alloc = [&](size_t bytes) -> void* {
    void* p = (void*)wp; wp += (bytes + 255) & ~(size_t)255; return p;
  };
  float* xd   = (float*)alloc((size_t)ND * 64 * 4);
  float* xp   = (float*)alloc((size_t)NP * 64 * 4);
  float* xp0  = (float*)alloc((size_t)NP * 64 * 4);
  float* s_p  = (float*)alloc((size_t)NP * 64 * 4);
  float* s_d  = (float*)alloc((size_t)ND * 64 * 4);
  float* g_p  = (float*)alloc((size_t)NP * 64 * 4);
  float* g_d  = (float*)alloc((size_t)ND * 64 * 4);
  float* agg  = (float*)alloc((size_t)ND * 64 * 4);
  float* deg  = (float*)alloc((size_t)ND * 4);
  float* hsrc = (float*)alloc((size_t)ND * 64 * 4);
  float* hdst = (float*)alloc((size_t)ND * 64 * 4);
  float* al_s = (float*)alloc((size_t)ND * 8 * 4);
  float* al_d = (float*)alloc((size_t)ND * 8 * 4);
  unsigned* mm = (unsigned*)alloc((size_t)ND * 8 * 4);
  float* den  = (float*)alloc((size_t)ND * 8 * 4);
  float* ebuf = (float*)alloc((size_t)EE * 8 * 4);
  float* pa   = (float*)alloc((size_t)NP * 256 * 4);
  float* pb   = (float*)alloc((size_t)NP * 256 * 4);

  // ---- input encoders + id embeddings ----
  GEMM(x_drug, W_dl, b_dl, xd, drug_id, emb_drug, ND, 64, 1024, 0);
  GEMM(x_prot, W_pl, b_pl, xp, prot_id, emb_prot, NP, 64, 1280, 0);
  hipMemcpyAsync(xp0, xp, (size_t)NP * 64 * 4, hipMemcpyDeviceToDevice, stream);

  // ---- two hetero layers ----
  for (int L = 0; L < 2; ++L) {
    const float* sWl = L ? s2_Wl : s1_Wl;
    const float* sbl = L ? s2_bl : s1_bl;
    const float* sWr = L ? s2_Wr : s1_Wr;
    const float* gW  = L ? g2_W  : g1_W;
    const float* gas = L ? g2_as : g1_as;
    const float* gad = L ? g2_ad : g1_ad;
    const float* gb  = L ? g2_b  : g1_b;
    const int heads = L ? 1 : 8;
    const int hsh   = L ? 0 : 3;   // log2(heads)
    const int csh   = L ? 6 : 3;   // log2(ch)
    const int ch    = L ? 64 : 8;

    // SAGE drug->protein (type 0)
    hipMemsetAsync(agg, 0, (size_t)NP * 64 * 4, stream);
    hipMemsetAsync(deg, 0, (size_t)NP * 4, stream);
    sage_agg_k<<<(EE + 7) / 8, 256, 0, stream>>>(xd, e_src, e_dst, EE, agg, deg);
    sage_mean_k EWG(NP * 64)(agg, deg, NP * 64);
    GEMM(agg, sWl, sbl, s_p, nullptr, nullptr, NP, 64, 64, 0);
    GEMM(xp, sWr, nullptr, s_p, nullptr, nullptr, NP, 64, 64, FLAG_ACC);

    // SAGE protein->drug (type 1, reversed edges)
    hipMemsetAsync(agg, 0, (size_t)ND * 64 * 4, stream);
    hipMemsetAsync(deg, 0, (size_t)ND * 4, stream);
    sage_agg_k<<<(EE + 7) / 8, 256, 0, stream>>>(xp, e_dst, e_src, EE, agg, deg);
    sage_mean_k EWG(ND * 64)(agg, deg, ND * 64);
    GEMM(agg, sWl + 4096, sbl + 64, s_d, nullptr, nullptr, ND, 64, 64, 0);
    GEMM(xd, sWr + 4096, nullptr, s_d, nullptr, nullptr, ND, 64, 64, FLAG_ACC);

    // GAT drug->protein (type 0)
    GEMM(xd, gW, nullptr, hsrc, nullptr, nullptr, ND, 64, 64, 0);
    GEMM(xp, gW, nullptr, hdst, nullptr, nullptr, NP, 64, 64, 0);
    att_reduce_k EWG(ND * heads)(hsrc, gas, al_s, ND * heads, hsh, ch);
    att_reduce_k EWG(NP * heads)(hdst, gad, al_d, NP * heads, hsh, ch);
    hipMemsetAsync(mm, 0, (size_t)NP * heads * 4, stream);
    hipMemsetAsync(den, 0, (size_t)NP * heads * 4, stream);
    edge_max_k EWG(EE * heads)(al_s, al_d, e_src, e_dst, EE, hsh, ebuf, mm);
    edge_expsum_k EWG(EE * heads)(ebuf, mm, den, e_dst, EE, hsh);
    hipMemsetAsync(g_p, 0, (size_t)NP * 64 * 4, stream);
    edge_agg_k<<<(EE + 7) / 8, 256, 0, stream>>>(hsrc, ebuf, den, e_src, e_dst,
                                                 EE, hsh, csh, g_p);

    // GAT protein->drug (type 1, reversed edges)
    GEMM(xp, gW + 4096, nullptr, hsrc, nullptr, nullptr, NP, 64, 64, 0);
    GEMM(xd, gW + 4096, nullptr, hdst, nullptr, nullptr, ND, 64, 64, 0);
    att_reduce_k EWG(NP * heads)(hsrc, gas + 64, al_s, NP * heads, hsh, ch);
    att_reduce_k EWG(ND * heads)(hdst, gad + 64, al_d, ND * heads, hsh, ch);
    hipMemsetAsync(mm, 0, (size_t)ND * heads * 4, stream);
    hipMemsetAsync(den, 0, (size_t)ND * heads * 4, stream);
    edge_max_k EWG(EE * heads)(al_s, al_d, e_dst, e_src, EE, hsh, ebuf, mm);
    edge_expsum_k EWG(EE * heads)(ebuf, mm, den, e_src, EE, hsh);
    hipMemsetAsync(g_d, 0, (size_t)ND * 64 * 4, stream);
    edge_agg_k<<<(EE + 7) / 8, 256, 0, stream>>>(hsrc, ebuf, den, e_dst, e_src,
                                                 EE, hsh, csh, g_d);

    // residual combine: x += relu(s)*sig + relu(g+b)*sig
    combine_k EWG(NP * 64)(xp, s_p, g_p, gb, NP * 64);
    combine_k EWG(ND * 64)(xd, s_d, g_d, gb + 64, ND * 64);
  }

  // ---- LTDTI protein head on xp0 ----
  GEMM(xp0, enc_W, enc_b, pa, nullptr, nullptr, NP, 64, 64, FLAG_LRELU);
  GEMM(pa, mha_Wv, mha_bv, pb, nullptr, nullptr, NP, 64, 64, 0);
  GEMM(pb, mha_Wo, mha_bo, hsrc, nullptr, nullptr, NP, 64, 64, 0);
  mix_k EWG(NP * 64)(pa, hsrc, NP * 64);
  maxpool_k EWG(NP * 32)(pa, pb, NP * 32);
  GEMM(pb, h_W1, h_b1, pa, nullptr, nullptr, NP, 256, 32, 0);
  GEMM(pa, h_W2, h_b2, pb, nullptr, nullptr, NP, 128, 256, FLAG_LRELU);
  GEMM(pb, h_W3, h_b3, pa, nullptr, nullptr, NP, 64, 128, FLAG_LRELU);
  GEMM(pa, o1_W, o1_b, pb, nullptr, nullptr, NP, 256, 64, 0);
  bn_lrelu_k EWG(NP * 256)(pb, bn1_g, bn1_b, NP * 256, 256);
  GEMM(pb, o2_W, o2_b, pa, nullptr, nullptr, NP, 128, 256, 0);
  bn_lrelu_k EWG(NP * 128)(pa, bn2_g, bn2_b, NP * 128, 128);
  GEMM(pa, o3_W, o3_b, pb, nullptr, nullptr, NP, 64, 128, 0);
  bn_lrelu_k EWG(NP * 64)(pb, bn3_g, bn3_b, NP * 64, 64);
  GEMM(pb, o4_W, o4_b, pa, nullptr, nullptr, NP, 64, 64, 0);
  sigmoid_k EWG(NP * 64)(pa, NP * 64);

  // ---- fused classifier ----
  classify_k<<<(EL + 31) / 32, 256, 0, stream>>>(xd, pa, l_src, l_dst,
                                                 c_W1, c_b1, c_W2, c_b2, out, EL);
}
